// AttentionBlock_20787641713164
// MI455X (gfx1250) — matrix operand order repaired and checked by reading, not compiled, not run on hardware
//
#include <hip/hip_runtime.h>

// ---------------------------------------------------------------------------
// CDNA5 (gfx1250) bf16 WMMA transformer block.
// - V_WMMA_F32_16X16X32_BF16 for all GEMMs (fp32 accumulate), wave32.
// - GLOBAL_LOAD_ASYNC_TO_LDS_B128 (ASYNCcnt) for all tile staging,
//   double-buffered in the GEMM k-loop.
// - Weights pre-transposed once (bf16 [N][K]); V written transposed by its
//   projection GEMM so attention staging is pure contiguous copies.
// - global_prefetch for the dominant attn_bias HBM stream (268 MB).
// ---------------------------------------------------------------------------

typedef __bf16 v16bf __attribute__((ext_vector_type(16)));
typedef float  v8f   __attribute__((ext_vector_type(8)));

union Frag { v16bf v; unsigned u[8]; unsigned short s[16]; };

static __device__ __forceinline__ unsigned short f2bf(float f) {
  unsigned u = __float_as_uint(f);
  u += 0x7fffu + ((u >> 16) & 1u);   // round-to-nearest-even
  return (unsigned short)(u >> 16);
}

static __device__ __forceinline__ v8f zero_v8() {
  v8f z;
#pragma unroll
  for (int i = 0; i < 8; ++i) z[i] = 0.f;
  return z;
}

// 16-bit A-matrix 16x32 K index for VGPR-pair p (0..7), lane-half h (0..1); contiguous k = 16h + 2p.
static __device__ __forceinline__ int a_k0(int p, int h) {
  return 2 * p + 16 * h;
}

static __device__ __forceinline__ v8f wmma_bf16(v16bf a, v16bf b, v8f c) {
  return __builtin_amdgcn_wmma_f32_16x16x32_bf16(
      false, a, false, b, (short)0, c, false, false);
}

// A fragment from a u16 row-major tile (rows = M, k contiguous), row stride ld.
static __device__ __forceinline__ v16bf frag_a(const unsigned short* t, int row0,
                                               int ld, int k0, int lane) {
  Frag f;
  const unsigned short* p = t + (size_t)(row0 + (lane & 15)) * ld + k0;
  int h = lane >> 4;
#pragma unroll
  for (int i = 0; i < 8; ++i) f.u[i] = *(const unsigned*)(p + a_k0(i, h));
  return f.v;
}

// B fragment from a u16 tile stored [n][k] (k contiguous per column n).
static __device__ __forceinline__ v16bf frag_b(const unsigned short* t, int col0,
                                               int ld, int k0, int lane) {
  Frag f;
  const unsigned short* p =
      t + (size_t)(col0 + (lane & 15)) * ld + k0 + 16 * (lane >> 4);
#pragma unroll
  for (int i = 0; i < 8; ++i) f.u[i] = *(const unsigned*)(p + 2 * i);
  return f.v;
}

// LDS byte offset of a __shared__ pointer (ISA 10.2: LDS aperture uses addr[31:0]).
static __device__ __forceinline__ unsigned lds_off(const void* p) {
  return (unsigned)(size_t)p;
}

// Async global->LDS 16B copy, tracked by ASYNCcnt (gfx1250 TDM-adjacent path).
static __device__ __forceinline__ void async_b128(unsigned lds, const void* g) {
  asm volatile("global_load_async_to_lds_b128 %0, %1, off"
               :: "v"(lds), "v"((unsigned long long)(size_t)g) : "memory");
}
static __device__ __forceinline__ void wait_async0() {
  asm volatile("s_wait_asynccnt 0x0" ::: "memory");
}

// ---------------------------------------------------------------------------
// Weight convert + transpose: fp32 src[K][N] -> bf16 dst[N][K]. 32x32 tiles.
// ---------------------------------------------------------------------------
__global__ __launch_bounds__(256) void cvt_t_kernel(const float* __restrict__ src,
                                                    unsigned short* __restrict__ dst,
                                                    int K, int N) {
  __shared__ float tile[32][33];
  int kb = blockIdx.y * 32, nb = blockIdx.x * 32;
  int tx = threadIdx.x & 31, ty = threadIdx.x >> 5;  // 32 x 8
#pragma unroll
  for (int i = ty; i < 32; i += 8)
    tile[i][tx] = src[(size_t)(kb + i) * N + nb + tx];
  __syncthreads();
#pragma unroll
  for (int i = ty; i < 32; i += 8)
    dst[(size_t)(nb + i) * K + kb + tx] = f2bf(tile[tx][i]);
}

// ---------------------------------------------------------------------------
// LayerNorm over H=1024, fp32 in -> bf16 out. One block (256 thr) per row.
// ---------------------------------------------------------------------------
__global__ __launch_bounds__(256) void ln_kernel(const float* __restrict__ x,
                                                 const float* __restrict__ g,
                                                 const float* __restrict__ bta,
                                                 unsigned short* __restrict__ y) {
  const int H = 1024;
  int row = blockIdx.x;
  const float* xr = x + (size_t)row * H;
  int t = threadIdx.x;
  float vals[4];
  float s = 0.f, s2 = 0.f;
#pragma unroll
  for (int i = 0; i < 4; ++i) {
    float v = xr[t + 256 * i];
    vals[i] = v;
    s += v;
    s2 += v * v;
  }
#pragma unroll
  for (int m = 16; m >= 1; m >>= 1) {
    s += __shfl_xor(s, m, 32);
    s2 += __shfl_xor(s2, m, 32);
  }
  __shared__ float rs[8], rs2[8];
  int w = t >> 5, lane = t & 31;
  if (lane == 0) { rs[w] = s; rs2[w] = s2; }
  __syncthreads();
  if (w == 0) {
    float a = (lane < 8) ? rs[lane] : 0.f;
    float a2 = (lane < 8) ? rs2[lane] : 0.f;
#pragma unroll
    for (int m = 4; m >= 1; m >>= 1) {
      a += __shfl_xor(a, m, 32);
      a2 += __shfl_xor(a2, m, 32);
    }
    if (lane == 0) { rs[0] = a; rs2[0] = a2; }
  }
  __syncthreads();
  float mean = rs[0] * (1.0f / H);
  float var = rs2[0] * (1.0f / H) - mean * mean;
  float inv = rsqrtf(var + 1e-5f);
  unsigned short* yr = y + (size_t)row * H;
#pragma unroll
  for (int i = 0; i < 4; ++i) {
    int c = t + 256 * i;
    yr[c] = f2bf((vals[i] - mean) * inv * g[c] + bta[c]);
  }
}

// ---------------------------------------------------------------------------
// Tiled bf16 GEMM: out[M,N] = A[M,K] @ Wt[N,K]^T + bias (+res) (+GELU)
// Wt is the pre-transposed weight (bf16, [N][K]).
// Block tile 128x128, 8 waves x (32x64). K staged 32-wide via async-to-LDS,
// double buffered so the next tile's loads overlap the current WMMAs.
// OUTMODE: 0 = f32, 1 = bf16, 2 = bf16 transposed to [B][NH][DK][S]
// (hardcoded S=2048, DK=64 for mode 2, used for the V projection).
// ---------------------------------------------------------------------------
template <int ACT, int OUTMODE, bool RES>
__global__ __launch_bounds__(256) void gemm_kernel(
    const unsigned short* __restrict__ A, const unsigned short* __restrict__ Wt,
    const float* __restrict__ bias, const float* __restrict__ res,
    void* __restrict__ outv, int M, int N, int K) {
  __shared__ unsigned short As[2][128 * 32];
  __shared__ unsigned short Bs[2][128 * 32];
  const int tid = threadIdx.x;
  const int lane = tid & 31, w = tid >> 5;
  const int wm = (w >> 1) * 32, wn = (w & 1) * 64;
  const int mb = blockIdx.y * 128, nb = blockIdx.x * 128;

  v8f acc[2][4];
#pragma unroll
  for (int i = 0; i < 2; ++i)
#pragma unroll
    for (int j = 0; j < 4; ++j) acc[i][j] = zero_v8();

  auto stage = [&](int buf, int kb) {
    unsigned aB = lds_off(&As[buf][0]);
    unsigned bB = lds_off(&Bs[buf][0]);
#pragma unroll
    for (int it = 0; it < 2; ++it) {
      int c = tid + it * 256;          // 512 x 16B chunks per tile (A and B each)
      int r = c >> 2, q = (c & 3) * 8; // row, u16 offset within 32-wide row
      async_b128(aB + (unsigned)(r * 64 + q * 2), A + (size_t)(mb + r) * K + kb + q);
      async_b128(bB + (unsigned)(r * 64 + q * 2), Wt + (size_t)(nb + r) * K + kb + q);
    }
  };

  stage(0, 0);
  wait_async0();
  __syncthreads();

  int buf = 0;
  for (int kb = 0; kb < K; kb += 32) {
    if (kb + 32 < K) stage(buf ^ 1, kb + 32);  // overlap with WMMAs below

    v16bf a0 = frag_a(As[buf], wm, 32, 0, lane);
    v16bf a1 = frag_a(As[buf], wm + 16, 32, 0, lane);
#pragma unroll
    for (int nf = 0; nf < 4; ++nf) {
      v16bf bf = frag_b(Bs[buf], wn + nf * 16, 32, 0, lane);
      acc[0][nf] = wmma_bf16(a0, bf, acc[0][nf]);
      acc[1][nf] = wmma_bf16(a1, bf, acc[1][nf]);
    }
    wait_async0();
    __syncthreads();
    buf ^= 1;
  }

  const int h2 = lane >> 4, cl = lane & 15;
  float* outF = (float*)outv;
  unsigned short* outB = (unsigned short*)outv;
#pragma unroll
  for (int mf = 0; mf < 2; ++mf)
#pragma unroll
    for (int nf = 0; nf < 4; ++nf) {
#pragma unroll
      for (int v = 0; v < 8; ++v) {
        int row = mb + wm + mf * 16 + v + 8 * h2;
        int col = nb + wn + nf * 16 + cl;
        float val = acc[mf][nf][v] + bias[col];
        if (ACT == 1) val = 0.5f * val * (1.0f + erff(val * 0.70710678118f));
        if (RES) val += res[(size_t)row * N + col];
        if (OUTMODE == 0) {
          outF[(size_t)row * N + col] = val;
        } else if (OUTMODE == 1) {
          outB[(size_t)row * N + col] = f2bf(val);
        } else {
          // V transposed: [B][NH][DK][S], S=2048, DK=64
          int bidx = row >> 11, sI = row & 2047;
          int hI = col >> 6, d = col & 63;
          outB[(((size_t)(bidx * 16 + hI) * 64 + d) << 11) + sI] = f2bf(val);
        }
      }
    }
}

// ---------------------------------------------------------------------------
// Flash attention: workgroup = (b, h, 128 query rows); each wave owns a
// 16-row strip with private online-softmax state. 64-key tiles staged via
// async-to-LDS; V comes pre-transposed ([B][NH][DK][S]); bias prefetched.
// ---------------------------------------------------------------------------
__global__ __launch_bounds__(256) void attn_kernel(
    const unsigned short* __restrict__ Q, const unsigned short* __restrict__ Kg,
    const unsigned short* __restrict__ VtG, const float* __restrict__ bias,
    unsigned short* __restrict__ Og) {
  const int S = 2048, H = 1024, DK = 64;
  const float SCALE = 0.125f;  // 64^-0.5
  __shared__ unsigned short Ks[64 * 64];      // [key][d]
  __shared__ unsigned short Vt[64 * 64];      // [d][key]
  __shared__ unsigned short Pb[8 * 16 * 64];  // per-wave P tiles
  const int tid = threadIdx.x, lane = tid & 31, w = tid >> 5;
  const int h2 = lane >> 4, cl = lane & 15;
  const int hh = blockIdx.y, b = blockIdx.z;
  const int qb = blockIdx.x * 128 + w * 16;

  // Q fragments (16 rows x DK=64 -> two 16x32 A-frags), kept in registers.
  const unsigned short* qp = Q + ((size_t)(b * S + qb + cl)) * H + hh * DK;
  Frag qa0, qa1;
#pragma unroll
  for (int i = 0; i < 8; ++i) qa0.u[i] = *(const unsigned*)(qp + a_k0(i, h2));
#pragma unroll
  for (int i = 0; i < 8; ++i) qa1.u[i] = *(const unsigned*)(qp + 32 + a_k0(i, h2));

  float m_i[8], l_i[8];
#pragma unroll
  for (int v = 0; v < 8; ++v) { m_i[v] = -1e30f; l_i[v] = 0.f; }
  v8f o[4];
#pragma unroll
  for (int nf = 0; nf < 4; ++nf) o[nf] = zero_v8();

  unsigned short* Pw = Pb + w * 16 * 64;
  const unsigned ksB = lds_off(&Ks[0]);
  const unsigned vtB = lds_off(&Vt[0]);
  const unsigned short* vrow = VtG + ((size_t)(b * 16 + hh) * 64) * S;

  for (int kb = 0; kb < S; kb += 64) {
    // stage K [key][d] and Vt [d][key] via async copies (contiguous rows)
#pragma unroll
    for (int it = 0; it < 2; ++it) {
      int c = tid + it * 256;            // 512 x 16B chunks each
      int r = c >> 3, q = (c & 7) * 8;   // row (key or d), u16 col offset
      async_b128(ksB + (unsigned)(r * 128 + q * 2),
                 Kg + ((size_t)(b * S + kb + r)) * H + hh * DK + q);
      async_b128(vtB + (unsigned)(r * 128 + q * 2),
                 vrow + (size_t)r * S + kb + q);
    }
    wait_async0();
    __syncthreads();

    // prefetch next tile's bias lines (dominant 268MB HBM stream)
    if (kb + 64 < S) {
#pragma unroll
      for (int nf = 0; nf < 4; ++nf)
#pragma unroll
        for (int v = 0; v < 8; ++v) {
          int qr = qb + v + 8 * h2;
          __builtin_prefetch(
              &bias[((size_t)hh * S + qr) * S + kb + 64 + nf * 16 + cl], 0, 1);
        }
    }

    // scores: S = Q . K^T  (contract over d; B columns are keys)
    v8f sc[4];
#pragma unroll
    for (int nf = 0; nf < 4; ++nf) {
      v16bf b0 = frag_b(Ks, nf * 16, 64, 0, lane);
      v8f c = wmma_bf16(qa0.v, b0, zero_v8());
      v16bf b1 = frag_b(Ks, nf * 16, 64, 32, lane);
      c = wmma_bf16(qa1.v, b1, c);
      sc[nf] = c;
    }
    // scale + attn_bias (coalesced over k)
#pragma unroll
    for (int nf = 0; nf < 4; ++nf) {
#pragma unroll
      for (int v = 0; v < 8; ++v) {
        int qr = qb + v + 8 * h2;
        int kc = kb + nf * 16 + cl;
        sc[nf][v] = sc[nf][v] * SCALE + bias[((size_t)hh * S + qr) * S + kc];
      }
    }
    // online softmax per row (rows split across half-waves; reduce width 16)
#pragma unroll
    for (int v = 0; v < 8; ++v) {
      float mx = fmaxf(fmaxf(sc[0][v], sc[1][v]), fmaxf(sc[2][v], sc[3][v]));
#pragma unroll
      for (int msk = 8; msk >= 1; msk >>= 1) mx = fmaxf(mx, __shfl_xor(mx, msk, 16));
      float mn = fmaxf(m_i[v], mx);
      float alpha = __expf(m_i[v] - mn);
      m_i[v] = mn;
      float rs = 0.f;
#pragma unroll
      for (int nf = 0; nf < 4; ++nf) {
        float p = __expf(sc[nf][v] - mn);
        sc[nf][v] = p;
        rs += p;
      }
#pragma unroll
      for (int msk = 8; msk >= 1; msk >>= 1) rs += __shfl_xor(rs, msk, 16);
      l_i[v] = l_i[v] * alpha + rs;
#pragma unroll
      for (int nf = 0; nf < 4; ++nf) o[nf][v] *= alpha;
    }
    // re-layout P (C-layout -> A-layout) through per-wave LDS tile
#pragma unroll
    for (int nf = 0; nf < 4; ++nf)
#pragma unroll
      for (int v = 0; v < 8; ++v)
        Pw[(v + 8 * h2) * 64 + nf * 16 + cl] = f2bf(sc[nf][v]);
    asm volatile("s_wait_dscnt 0x0" ::: "memory");

    v16bf p0 = frag_a(Pw, 0, 64, 0, lane);
    v16bf p1 = frag_a(Pw, 0, 64, 32, lane);
#pragma unroll
    for (int nf = 0; nf < 4; ++nf) {
      v16bf v0 = frag_b(Vt, nf * 16, 64, 0, lane);
      o[nf] = wmma_bf16(p0, v0, o[nf]);
      v16bf v1 = frag_b(Vt, nf * 16, 64, 32, lane);
      o[nf] = wmma_bf16(p1, v1, o[nf]);
    }
    __syncthreads();
  }

  // normalize and store bf16 attention output [B,S,H]
#pragma unroll
  for (int v = 0; v < 8; ++v) l_i[v] = 1.0f / l_i[v];
#pragma unroll
  for (int nf = 0; nf < 4; ++nf)
#pragma unroll
    for (int v = 0; v < 8; ++v) {
      int qr = qb + v + 8 * h2;
      int d = nf * 16 + cl;
      Og[((size_t)(b * S + qr)) * H + hh * DK + d] = f2bf(o[nf][v] * l_i[v]);
    }
}

// ---------------------------------------------------------------------------
// Host-side orchestration
// ---------------------------------------------------------------------------
extern "C" void kernel_launch(void* const* d_in, const int* in_sizes, int n_in,
                              void* d_out, int out_size, void* d_ws, size_t ws_size,
                              hipStream_t stream) {
  const int B = 4, S = 2048, H = 1024, FFN = 4096;
  const int M = B * S;

  const float* x    = (const float*)d_in[0];
  const float* bias = (const float*)d_in[1];
  const float* ln1g = (const float*)d_in[2];
  const float* ln1b = (const float*)d_in[3];
  const float* Wq   = (const float*)d_in[4];
  const float* bq   = (const float*)d_in[5];
  const float* Wk   = (const float*)d_in[6];
  const float* bk   = (const float*)d_in[7];
  const float* Wv   = (const float*)d_in[8];
  const float* bv   = (const float*)d_in[9];
  const float* Wo   = (const float*)d_in[10];
  const float* bo   = (const float*)d_in[11];
  const float* ln2g = (const float*)d_in[12];
  const float* ln2b = (const float*)d_in[13];
  const float* W1   = (const float*)d_in[14];
  const float* b1   = (const float*)d_in[15];
  const float* W2   = (const float*)d_in[16];
  const float* b2   = (const float*)d_in[17];

  char* ws = (char*)d_ws;
  size_t off = 0;
  auto take = [&](size_t bytes) -> void* {
    void* p = ws + off;
    off += (bytes + 255) & ~(size_t)255;
    return p;
  };
  unsigned short* WqT = (unsigned short*)take((size_t)H * H * 2);
  unsigned short* WkT = (unsigned short*)take((size_t)H * H * 2);
  unsigned short* WvT = (unsigned short*)take((size_t)H * H * 2);
  unsigned short* WoT = (unsigned short*)take((size_t)H * H * 2);
  unsigned short* W1T = (unsigned short*)take((size_t)H * FFN * 2);
  unsigned short* W2T = (unsigned short*)take((size_t)FFN * H * 2);
  unsigned short* y1  = (unsigned short*)take((size_t)M * H * 2);
  unsigned short* Qb  = (unsigned short*)take((size_t)M * H * 2);
  unsigned short* Kb  = (unsigned short*)take((size_t)M * H * 2);
  unsigned short* VtG = (unsigned short*)take((size_t)M * H * 2);  // [B][NH][DK][S]
  unsigned short* AOb = (unsigned short*)take((size_t)M * H * 2);
  float*          x2  = (float*)take((size_t)M * H * 4);
  unsigned short* y2  = (unsigned short*)take((size_t)M * H * 2);
  unsigned short* h1  = (unsigned short*)take((size_t)M * FFN * 2);

  dim3 blk(256);
  auto cvt_t = [&](const float* s, unsigned short* d, int K, int N) {
    cvt_t_kernel<<<dim3(N / 32, K / 32), blk, 0, stream>>>(s, d, K, N);
  };
  cvt_t(Wq, WqT, H, H);
  cvt_t(Wk, WkT, H, H);
  cvt_t(Wv, WvT, H, H);
  cvt_t(Wo, WoT, H, H);
  cvt_t(W1, W1T, H, FFN);
  cvt_t(W2, W2T, FFN, H);

  // y1 = LN1(x)
  ln_kernel<<<dim3(M), blk, 0, stream>>>(x, ln1g, ln1b, y1);

  // Q,K projections (bf16); V projection written transposed [B][NH][DK][S]
  gemm_kernel<0, 1, false><<<dim3(H / 128, M / 128), blk, 0, stream>>>(
      y1, WqT, bq, nullptr, Qb, M, H, H);
  gemm_kernel<0, 1, false><<<dim3(H / 128, M / 128), blk, 0, stream>>>(
      y1, WkT, bk, nullptr, Kb, M, H, H);
  gemm_kernel<0, 2, false><<<dim3(H / 128, M / 128), blk, 0, stream>>>(
      y1, WvT, bv, nullptr, VtG, M, H, H);

  // flash attention with bias
  attn_kernel<<<dim3(S / 128, 16, B), blk, 0, stream>>>(Qb, Kb, VtG, bias, AOb);

  // x2 = x + attn_out @ Wo + bo   (f32 out)
  gemm_kernel<0, 0, true><<<dim3(H / 128, M / 128), blk, 0, stream>>>(
      AOb, WoT, bo, x, x2, M, H, H);

  // y2 = LN2(x2)
  ln_kernel<<<dim3(M), blk, 0, stream>>>(x2, ln2g, ln2b, y2);

  // h1 = GELU(y2 @ W1 + b1)  (bf16 out)
  gemm_kernel<1, 1, false><<<dim3(FFN / 128, M / 128), blk, 0, stream>>>(
      y2, W1T, b1, nullptr, h1, M, FFN, H);

  // out = x2 + h1 @ W2 + b2  (f32 out, final)
  gemm_kernel<0, 0, true><<<dim3(H / 128, M / 128), blk, 0, stream>>>(
      h1, W2T, b2, x2, (float*)d_out, M, H, FFN);
}
